// LongformerSelfAttention_25658134627018
// MI455X (gfx1250) — matrix operand order repaired and checked by reading, not compiled, not run on hardware
//
#include <hip/hip_runtime.h>
#include <stdint.h>

#define BATCH 2
#define SEQ   4096
#define EMB   512
#define NH    8
#define HD    64
#define WIN   64
#define NC    64      /* SEQ / WIN */
#define Y3    192     /* 3 * WIN   */
#define SLD   196     /* padded LDS score row stride (floats): 8*196 % 64 == 32 -> conflict-free */

typedef __bf16 bf16_t;
typedef __attribute__((ext_vector_type(16))) __bf16 bf16x16;
typedef __attribute__((ext_vector_type(8)))  __bf16 bf16x8;
typedef __attribute__((ext_vector_type(8)))  float  v8f;

union Frag16 {
  bf16x16  v;
  bf16x8   h[2];
  uint32_t u[8];
};

__device__ __forceinline__ v8f wmma_bf16(bf16x16 a, bf16x16 b, v8f c) {
  // D(f32 16x16) = A(16x32 bf16) * B(32x16 bf16) + C
  return __builtin_amdgcn_wmma_f32_16x16x32_bf16(
      /*neg_a=*/false, a, /*neg_b=*/false, b,
      /*c_mod=*/(short)0, c, /*reuse_a=*/false, /*reuse_b=*/false);
}

// ---------------------------------------------------------------------------
// Kernel 0a: f32 -> bf16 convert of hidden_states
// ---------------------------------------------------------------------------
__global__ void cvt_hidden(const float* __restrict__ src, bf16_t* __restrict__ dst, int n) {
  int i = blockIdx.x * blockDim.x + threadIdx.x;
  if (i < n) dst[i] = (bf16_t)src[i];
}

// ---------------------------------------------------------------------------
// Kernel 0b: f32 weights -> bf16, transposed (Wt[z][n*E + k] = W_z[k*E + n])
// so GEMM B-fragments read contiguous K per output column.
// ---------------------------------------------------------------------------
__global__ void cvt_weights(const float* __restrict__ Wq, const float* __restrict__ Wk,
                            const float* __restrict__ Wv, bf16_t* __restrict__ Wt) {
  int i = blockIdx.x * blockDim.x + threadIdx.x;   // i in [0, 3*E*E)
  int z   = i >> 18;                               // E*E = 262144 = 2^18
  int rem = i & (262144 - 1);
  int n   = rem >> 9;                              // E = 512 = 2^9
  int k   = rem & 511;
  const float* Wsrc = (z == 0) ? Wq : ((z == 1) ? Wk : Wv);
  Wt[(size_t)z * 262144 + (size_t)n * EMB + k] = (bf16_t)Wsrc[(size_t)k * EMB + n];
}

// ---------------------------------------------------------------------------
// Kernel 1: QKV projection GEMM, bf16 WMMA, f32 accumulate.
//   grid = (M/64, E/64, 3); block = 128 (4 waves); wave w -> 16-row strip.
//   Software-pipelined: double-buffered A/B fragments + global_prefetch of
//   upcoming K-tiles so WMMAs overlap with loads.
//   z==0: q (scaled 1/sqrt(D)) -> qb[(b*S+s)*E + h*64+d]   (bf16)
//   z==1: k                    -> kb same layout            (bf16)
//   z==2: v stored transposed  -> vt[((b*H+h)*HD+d)*S + s]  (bf16), packed 16B stores
// ---------------------------------------------------------------------------
__global__ __launch_bounds__(128)
void qkv_gemm(const bf16_t* __restrict__ Xb, const bf16_t* __restrict__ Wt,
              const float* __restrict__ bq, const float* __restrict__ bk,
              const float* __restrict__ bv,
              bf16_t* __restrict__ qb, bf16_t* __restrict__ kb, bf16_t* __restrict__ vt) {
  const int z    = blockIdx.z;
  const int lane = threadIdx.x & 31;
  const int wave = threadIdx.x >> 5;
  const int grp  = lane >> 4;       // which 16-lane half
  const int lidx = lane & 15;
  const int m0   = blockIdx.x * 64 + wave * 16;
  const int n0   = blockIdx.y * 64;

  const bf16_t* Wm   = Wt + (size_t)z * EMB * EMB;
  const float*  bias = (z == 0) ? bq : ((z == 1) ? bk : bv);

  v8f acc[4];
#pragma unroll
  for (int t = 0; t < 4; t++)
#pragma unroll
    for (int r = 0; r < 8; r++) acc[t][r] = 0.0f;

  const int     arow  = m0 + lidx;
  const bf16_t* abase = Xb + (size_t)arow * EMB;

  // Double-buffered fragment registers.
  Frag16 A[2];
  Frag16 Bf[2][4];

  auto load_tile = [&](int k0, int buf) {
    const bf16_t* ap = abase + k0 + grp * 8;     // A: row=lidx; K grp*8..+7, 16+grp*8..+7
    A[buf].h[0] = *(const bf16x8*)(ap);
    A[buf].h[1] = *(const bf16x8*)(ap + 16);
#pragma unroll
    for (int t = 0; t < 4; t++) {                // B: col=lidx; K grp*8..+7, 16+grp*8..+7
      const bf16_t* bp = Wm + (size_t)(n0 + t * 16 + lidx) * EMB + k0 + grp * 8;
      Bf[buf][t].h[0] = *(const bf16x8*)(bp);
      Bf[buf][t].h[1] = *(const bf16x8*)(bp + 16);
    }
  };

  load_tile(0, 0);
#pragma unroll
  for (int k0 = 0; k0 < EMB; k0 += 32) {
    const int cur = (k0 >> 5) & 1;
    if (k0 + 32 < EMB) {
      load_tile(k0 + 32, cur ^ 1);
      if (k0 + 64 < EMB) {
        // pull the tile after next toward the WGP (global_prefetch_b8)
        __builtin_prefetch(abase + k0 + 64, 0, 0);
        __builtin_prefetch(Wm + (size_t)(n0 + lidx) * EMB + k0 + 64, 0, 0);
      }
    }
#pragma unroll
    for (int t = 0; t < 4; t++) acc[t] = wmma_bf16(A[cur].v, Bf[cur][t].v, acc[t]);
  }

  if (z == 2) {
    // v: 8 consecutive s per lane (fixed d) -> one packed 16B store per tile
    const int m = m0 + 8 * grp;                  // first of this lane's 8 rows
    const int b = m >> 12, s = m & (SEQ - 1);    // rows never cross batch (64 | 4096)
#pragma unroll
    for (int t = 0; t < 4; t++) {
      const int   n  = n0 + t * 16 + lidx;
      const float bi = bias[n];
      const int   h  = n >> 6, d = n & 63;
      bf16x8 pk;
#pragma unroll
      for (int r = 0; r < 8; r++) pk[r] = (bf16_t)(acc[t][r] + bi);
      *(bf16x8*)(vt + (((size_t)b * NH + h) * HD + d) * SEQ + s) = pk;
    }
  } else {
    const float scale = (z == 0) ? 0.125f : 1.0f;   // q: 1/sqrt(64)
    bf16_t* dst = (z == 0) ? qb : kb;
#pragma unroll
    for (int t = 0; t < 4; t++) {
      const int   n  = n0 + t * 16 + lidx;
      const float bi = bias[n];
#pragma unroll
      for (int r = 0; r < 8; r++) {
        const int m = m0 + r + 8 * grp;          // C layout: VGPR r -> row r + 8*grp
        dst[(size_t)m * EMB + n] = (bf16_t)((acc[t][r] + bi) * scale);
      }
    }
  }
}

// ---------------------------------------------------------------------------
// Kernel 2: banded attention per (b, c, h). block = 128 threads (4 waves).
// ---------------------------------------------------------------------------
__global__ __launch_bounds__(128)
void attn(const bf16_t* __restrict__ qb, const bf16_t* __restrict__ kb,
          const bf16_t* __restrict__ vt, const int* __restrict__ amask,
          float* __restrict__ out) {
  __shared__ float sc[64 * SLD];   // 50176 bytes

  const int c    = blockIdx.x;
  const int h    = blockIdx.y;
  const int b    = blockIdx.z;
  const int lane = threadIdx.x & 31;
  const int wave = threadIdx.x >> 5;
  const int grp  = lane >> 4;
  const int lidx = lane & 15;
  const int xs   = wave * 16;      // this wave's 16 query rows

  // Prefetch this block's V strips (used in phase 4) while QK^T runs.
  {
    const int     d0    = lidx;
    const bf16_t* vbase = vt + (((size_t)(b * NH + h)) * HD + d0) * SEQ;
    const int     sbeg  = c * WIN - WIN;
    if (sbeg >= 0 && sbeg < SEQ) __builtin_prefetch(vbase + sbeg, 0, 0);
  }

  // ---- Phase 1: scores = q @ k^T  (64 x 192), 12 y-tiles x 2 K-steps ----
  v8f acc[12];
#pragma unroll
  for (int t = 0; t < 12; t++)
#pragma unroll
    for (int r = 0; r < 8; r++) acc[t][r] = 0.0f;

  const int     qrow  = c * WIN + xs + lidx;
  const bf16_t* qbase = qb + ((size_t)(b * SEQ + qrow)) * EMB + h * HD;
#pragma unroll
  for (int ks = 0; ks < HD; ks += 32) {
    Frag16 A;
    A.h[0] = *(const bf16x8*)(qbase + ks + grp * 8);
    A.h[1] = *(const bf16x8*)(qbase + ks + grp * 8 + 16);
#pragma unroll
    for (int yt = 0; yt < 12; yt++) {
      const int y  = yt * 16 + lidx;
      const int sg = c * WIN + y - WIN;          // global key position
      Frag16 Bf;
      if (sg >= 0 && sg < SEQ) {
        const bf16_t* kp = kb + ((size_t)(b * SEQ + sg)) * EMB + h * HD + ks + grp * 8;
        Bf.h[0] = *(const bf16x8*)(kp);
        Bf.h[1] = *(const bf16x8*)(kp + 16);
      } else {
#pragma unroll
        for (int i = 0; i < 8; i++) Bf.u[i] = 0u;
      }
      acc[yt] = wmma_bf16(A.v, Bf.v, acc[yt]);
    }
  }

  // ---- Phase 2: band mask + attention mask, spill to LDS ----
#pragma unroll
  for (int yt = 0; yt < 12; yt++) {
    const int  y   = yt * 16 + lidx;
    const int  sg  = c * WIN + y - WIN;
    const bool inr = (sg >= 0) && (sg < SEQ);
    float mv = 0.0f;
    if (inr && amask[b * SEQ + sg] != 0) mv = -10000.0f;
#pragma unroll
    for (int r = 0; r < 8; r++) {
      const int  x    = xs + r + 8 * grp;
      const bool band = (y >= x) && (y <= x + 2 * WIN);
      sc[x * SLD + y] = (band && inr) ? (acc[yt][r] + mv) : -__builtin_inff();
    }
  }
  __syncthreads();

  // ---- Phase 3: row-wise softmax in LDS (normalized probs stored in place) --
  if (threadIdx.x < 64) {
    const int x   = threadIdx.x;
    float*    row = sc + x * SLD;
    float mx = -__builtin_inff();
    for (int y = 0; y < Y3; y++) mx = fmaxf(mx, row[y]);
    float sum = 0.0f;
    for (int y = 0; y < Y3; y++) { float e = __expf(row[y] - mx); row[y] = e; sum += e; }
    const float keep = (amask[b * SEQ + c * WIN + x] >= 0) ? 1.0f : 0.0f;
    const float inv  = keep / sum;
    for (int y = 0; y < Y3; y++) row[y] *= inv;
  }
  __syncthreads();

  // ---- Phase 4: out = probs @ v  (64 x 64), K = 192 ----
  v8f oacc[4];
#pragma unroll
  for (int t = 0; t < 4; t++)
#pragma unroll
    for (int r = 0; r < 8; r++) oacc[t][r] = 0.0f;

  const float* srow = sc + (xs + lidx) * SLD;
#pragma unroll
  for (int ks = 0; ks < Y3; ks += 32) {
    Frag16 A;   // convert probs f32 (LDS) -> bf16 fragment in registers
#pragma unroll
    for (int i = 0; i < 8; i++) A.h[0][i] = (bf16_t)srow[ks + grp * 16 + i];
#pragma unroll
    for (int i = 0; i < 8; i++) A.h[1][i] = (bf16_t)srow[ks + grp * 16 + 8 + i];
#pragma unroll
    for (int nt = 0; nt < 4; nt++) {
      const int     d     = nt * 16 + lidx;
      const bf16_t* vbase = vt + (((size_t)(b * NH + h)) * HD + d) * SEQ + (c * WIN - WIN);
      const int y0 = ks + grp * 16;               // 16 contiguous K (= key pos) values
      const int s0 = c * WIN - WIN + y0;          // chunk aligned: all-in or all-out
      Frag16 Bf;
      if (s0 >= 0 && s0 < SEQ) {
        Bf.h[0] = *(const bf16x8*)(vbase + y0);
        Bf.h[1] = *(const bf16x8*)(vbase + y0 + 8);
      } else {
#pragma unroll
        for (int i = 0; i < 8; i++) Bf.u[i] = 0u;
      }
      oacc[nt] = wmma_bf16(A.v, Bf.v, oacc[nt]);
    }
  }

  // ---- Phase 5: store f32 output (B, S, E) ----
#pragma unroll
  for (int nt = 0; nt < 4; nt++) {
    const int d = nt * 16 + lidx;
#pragma unroll
    for (int r = 0; r < 8; r++) {
      const int x = xs + r + 8 * grp;
      out[((size_t)(b * SEQ + c * WIN + x)) * EMB + h * HD + d] = oacc[nt][r];
    }
  }
}

// ---------------------------------------------------------------------------
// Host launcher
// ---------------------------------------------------------------------------
extern "C" void kernel_launch(void* const* d_in, const int* in_sizes, int n_in,
                              void* d_out, int out_size, void* d_ws, size_t ws_size,
                              hipStream_t stream) {
  const float* hidden = (const float*)d_in[0];
  const int*   amask  = (const int*)d_in[3];
  const float* Wq     = (const float*)d_in[4];
  const float* bq     = (const float*)d_in[5];
  const float* Wk     = (const float*)d_in[6];
  const float* bk     = (const float*)d_in[7];
  const float* Wv     = (const float*)d_in[8];
  const float* bv     = (const float*)d_in[9];
  float*       out    = (float*)d_out;

  // workspace layout (bytes)
  char*   ws  = (char*)d_ws;
  bf16_t* Xb  = (bf16_t*)(ws);                 // B*S*E bf16      = 8388608
  bf16_t* Wt  = (bf16_t*)(ws + 8388608);       // 3*E*E bf16      = 1572864
  bf16_t* qbf = (bf16_t*)(ws + 9961472);       // B*S*E bf16
  bf16_t* kbf = (bf16_t*)(ws + 18350080);      // B*S*E bf16
  bf16_t* vtb = (bf16_t*)(ws + 26738688);      // B*H*D*S bf16 (v transposed)

  const int nh = BATCH * SEQ * EMB;            // 4194304
  cvt_hidden<<<(nh + 255) / 256, 256, 0, stream>>>(hidden, Xb, nh);
  cvt_weights<<<(3 * EMB * EMB) / 256, 256, 0, stream>>>(Wq, Wk, Wv, Wt);

  dim3 g1(BATCH * SEQ / 64, EMB / 64, 3);      // (128, 8, 3)
  qkv_gemm<<<g1, 128, 0, stream>>>(Xb, Wt, bq, bk, bv, qbf, kbf, vtb);

  dim3 g2(NC, NH, BATCH);                      // (64, 8, 2)
  attn<<<g2, 128, 0, stream>>>(qbf, kbf, vtb, amask, out);
}
